// EngineAttention_8392366096386
// MI455X (gfx1250) — matrix operand order repaired and checked by reading, not compiled, not run on hardware
//
#include <hip/hip_runtime.h>

// ---------------- problem constants ----------------
#define Bc    4
#define Tc    2048
#define DINc  1024
#define DOUTc 1024
#define NHc   16
#define HDc   64
#define QKVLD (3 * DOUTc)          // 3072
#define SCALEc 0.125f              // 1/sqrt(64)

// ---------------- vector types for WMMA ----------------
typedef __attribute__((ext_vector_type(16))) __bf16 v16bf;
typedef __attribute__((ext_vector_type(8)))  __bf16 v8bf;
typedef __attribute__((ext_vector_type(8)))  float  v8f;

union FragU { v16bf v; v8bf h[2]; };

static __device__ __forceinline__ unsigned short f2bf(float f) {
  unsigned int u = __float_as_uint(f);
  u += 0x7FFFu + ((u >> 16) & 1u);      // round-to-nearest-even
  return (unsigned short)(u >> 16);
}

static __device__ __forceinline__ v8f zero8() {
  v8f z;
#pragma unroll
  for (int i = 0; i < 8; ++i) z[i] = 0.0f;
  return z;
}

// A-fragment (16x32 bf16): lane L holds row M=L&15.
// elems 0..7  -> K = (L>>4)*8 .. +7 ; elems 8..15 -> K = 16+(L>>4)*8 .. +7
static __device__ __forceinline__ v16bf load_frag_a(const unsigned short* __restrict__ p,
                                                    size_t ld) {
  const int lane = threadIdx.x & 31;
  const int hb = lane >> 4, m = lane & 15;
  const unsigned short* row = p + (size_t)m * ld;
  FragU f;
  f.h[0] = *(const v8bf*)(row + hb * 8);
  f.h[1] = *(const v8bf*)(row + 16 + hb * 8);
  return f.v;
}

// B-fragment (32x16 bf16): lane L holds col N=L&15, elems 0..7 -> K = (L>>4)*8 .. +7 ; elems 8..15 -> K = 16+(L>>4)*8 .. +7
// p points at (row n0, col k0) of a K-contiguous [N,K] matrix.
static __device__ __forceinline__ v16bf load_frag_b(const unsigned short* __restrict__ p,
                                                    size_t ld) {
  const int lane = threadIdx.x & 31;
  const int hb = lane >> 4, n = lane & 15;
  const unsigned short* row = p + (size_t)n * ld;
  FragU f;
  f.h[0] = *(const v8bf*)(row + hb * 8);
  f.h[1] = *(const v8bf*)(row + 16 + hb * 8);
  return f.v;
}

#define WMMA_BF16(a, b, c) \
  __builtin_amdgcn_wmma_f32_16x16x32_bf16(false, (a), false, (b), (short)0, (c), false, false)

// ---------------- kernel 1: fp32 -> bf16 convert ----------------
__global__ void cvt_f32_bf16(const float* __restrict__ src,
                             unsigned short* __restrict__ dst, size_t n) {
  size_t i = (size_t)blockIdx.x * blockDim.x + threadIdx.x;
  size_t stride = (size_t)gridDim.x * blockDim.x;
  for (; i < n; i += stride) dst[i] = f2bf(src[i]);
}

// ---------------- kernel 2/5: 64x64-per-wave WMMA GEMM ----------------
// out[m][n] = sum_k A[m][k] * W[n][k]   (both K-contiguous row-major)
template <bool F32OUT>
__global__ void gemm64x64(const unsigned short* __restrict__ A,
                          const unsigned short* __restrict__ W,
                          void* __restrict__ out, int M, int N, int K) {
  const int wid = (int)((blockIdx.x * blockDim.x + threadIdx.x) >> 5);
  const int nt = N >> 6;
  const int m0 = (wid / nt) << 6;
  const int n0 = (wid % nt) << 6;
  if (m0 >= M) return;
  const int lane = threadIdx.x & 31;
  const int hb = lane >> 4, l15 = lane & 15;

  v8f acc[4][4];
#pragma unroll
  for (int i = 0; i < 4; ++i)
#pragma unroll
    for (int j = 0; j < 4; ++j) acc[i][j] = zero8();

  for (int k0 = 0; k0 < K; k0 += 32) {
    v16bf af[4], bfr[4];
#pragma unroll
    for (int i = 0; i < 4; ++i)
      af[i] = load_frag_a(A + (size_t)(m0 + 16 * i) * K + k0, K);
#pragma unroll
    for (int j = 0; j < 4; ++j)
      bfr[j] = load_frag_b(W + (size_t)(n0 + 16 * j) * K + k0, K);
#pragma unroll
    for (int i = 0; i < 4; ++i)
#pragma unroll
      for (int j = 0; j < 4; ++j)
        acc[i][j] = WMMA_BF16(af[i], bfr[j], acc[i][j]);
  }

#pragma unroll
  for (int i = 0; i < 4; ++i)
#pragma unroll
    for (int j = 0; j < 4; ++j)
#pragma unroll
      for (int r = 0; r < 8; ++r) {
        const int row = m0 + 16 * i + r + 8 * hb;
        const int col = n0 + 16 * j + l15;
        if (F32OUT)
          ((float*)out)[(size_t)row * N + col] = acc[i][j][r];
        else
          ((unsigned short*)out)[(size_t)row * N + col] = f2bf(acc[i][j][r]);
      }
}

// ---------------- kernel 3: V -> V^T  ([b,h,d,t]) ----------------
__global__ void transpose_v(const unsigned short* __restrict__ qkv,
                            unsigned short* __restrict__ vt) {
  const size_t n = (size_t)Bc * NHc * HDc * Tc;
  size_t i = (size_t)blockIdx.x * blockDim.x + threadIdx.x;
  size_t stride = (size_t)gridDim.x * blockDim.x;
  for (; i < n; i += stride) {
    const int t = (int)(i % Tc);
    size_t r = i / Tc;
    const int d = (int)(r % HDc);
    r /= HDc;
    const int h = (int)(r % NHc);
    const int b = (int)(r / NHc);
    vt[i] = qkv[((size_t)b * Tc + t) * QKVLD + 2 * DOUTc + h * HDc + d];
  }
}

// ---------------- kernel 4: flash-style causal attention ----------------
// one wave = 16 queries x full head (HD=64); key blocks of 32.
__global__ void attn_kernel(const unsigned short* __restrict__ qkv,
                            const unsigned short* __restrict__ vt,
                            unsigned short* __restrict__ ctx) {
  __shared__ unsigned short pbuf[8][16 * 32];   // per-wave P staging (1 KB each)

  const int widb = threadIdx.x >> 5;
  const int lane = threadIdx.x & 31;
  const int hb = lane >> 4, l15 = lane & 15;

  const int gwave = blockIdx.x * (blockDim.x >> 5) + widb;
  const int QB = Tc / 16;
  const int qb = gwave % QB;
  const int bh = gwave / QB;
  const int h = bh % NHc;
  const int b = bh / NHc;
  const int q0 = qb * 16;

  const unsigned short* Qb = qkv + ((size_t)b * Tc) * QKVLD + (size_t)h * HDc;
  const unsigned short* Kb = Qb + DOUTc;
  const unsigned short* Vb = vt + ((size_t)(b * NHc + h) * HDc) * Tc;

  // Q fragments for the two 32-wide K-dim steps (HD = 64)
  const v16bf aq0 = load_frag_a(Qb + (size_t)q0 * QKVLD, QKVLD);
  const v16bf aq1 = load_frag_a(Qb + (size_t)q0 * QKVLD + 32, QKVLD);

  v8f O[4];
#pragma unroll
  for (int dt = 0; dt < 4; ++dt) O[dt] = zero8();
  float mrow[8], lrow[8];
#pragma unroll
  for (int r = 0; r < 8; ++r) { mrow[r] = -3.0e38f; lrow[r] = 0.0f; }

  const int nkb = (q0 + 47) >> 5;   // covers keys 0 .. q0+15
  for (int jb = 0; jb < nkb; ++jb) {
    const int j0 = jb * 32;

    // ---- scores S = Q K^T for 2 sub-tiles of 16 keys ----
    float sv[2][8];
#pragma unroll
    for (int sub = 0; sub < 2; ++sub) {
      const int kbase = j0 + sub * 16;
      v16bf bk0 = load_frag_b(Kb + (size_t)kbase * QKVLD, QKVLD);
      v16bf bk1 = load_frag_b(Kb + (size_t)kbase * QKVLD + 32, QKVLD);
      v8f s = zero8();
      s = WMMA_BF16(aq0, bk0, s);
      s = WMMA_BF16(aq1, bk1, s);
#pragma unroll
      for (int r = 0; r < 8; ++r) {
        const int qg = q0 + r + 8 * hb;
        const int kg = kbase + l15;
        sv[sub][r] = s[r] * SCALEc + ((kg > qg) ? -10000.0f : 0.0f);
      }
    }

    // ---- online softmax (row = query; reduce across 16-lane half-group) ----
    float alpha[8], psum[8];
#pragma unroll
    for (int r = 0; r < 8; ++r) {
      float v = fmaxf(sv[0][r], sv[1][r]);
#pragma unroll
      for (int off = 1; off < 16; off <<= 1) v = fmaxf(v, __shfl_xor(v, off, 16));
      const float mnew = fmaxf(mrow[r], v);
      alpha[r] = __expf(mrow[r] - mnew);
      mrow[r] = mnew;
      float p0 = __expf(sv[0][r] - mnew);
      float p1 = __expf(sv[1][r] - mnew);
      pbuf[widb][(r + 8 * hb) * 32 + l15]      = f2bf(p0);
      pbuf[widb][(r + 8 * hb) * 32 + 16 + l15] = f2bf(p1);
      psum[r] = p0 + p1;
    }
#pragma unroll
    for (int r = 0; r < 8; ++r) {
      float s = psum[r];
#pragma unroll
      for (int off = 1; off < 16; off <<= 1) s += __shfl_xor(s, off, 16);
      lrow[r] = lrow[r] * alpha[r] + s;
    }
#pragma unroll
    for (int dt = 0; dt < 4; ++dt)
#pragma unroll
      for (int r = 0; r < 8; ++r) O[dt][r] *= alpha[r];

    // ---- reload P in A-fragment layout from LDS ----
    const unsigned short* base = &pbuf[widb][(size_t)l15 * 32];
    FragU fp;
    fp.h[0] = *(const v8bf*)(base + hb * 8);
    fp.h[1] = *(const v8bf*)(base + 16 + hb * 8);

    // ---- O += P * V  (V^T is [d][t] so B loads are contiguous) ----
#pragma unroll
    for (int dt = 0; dt < 4; ++dt) {
      v16bf bv = load_frag_b(Vb + (size_t)(dt * 16) * Tc + j0, Tc);
      O[dt] = WMMA_BF16(fp.v, bv, O[dt]);
    }
  }

  // ---- normalize + write context (bf16) ----
#pragma unroll
  for (int dt = 0; dt < 4; ++dt)
#pragma unroll
    for (int r = 0; r < 8; ++r) {
      const int qg = q0 + r + 8 * hb;
      const float o = O[dt][r] / lrow[r];
      ctx[((size_t)b * Tc + qg) * DOUTc + h * HDc + dt * 16 + l15] = f2bf(o);
    }
}

// ---------------- launcher ----------------
extern "C" void kernel_launch(void* const* d_in, const int* in_sizes, int n_in,
                              void* d_out, int out_size, void* d_ws, size_t ws_size,
                              hipStream_t stream) {
  (void)in_sizes; (void)n_in; (void)out_size; (void)ws_size;
  const float* x     = (const float*)d_in[0];
  const float* w_qkv = (const float*)d_in[1];
  const float* w_out = (const float*)d_in[2];
  float* out = (float*)d_out;

  const size_t nX    = (size_t)Bc * Tc * DINc;          // 8,388,608
  const size_t nWqkv = (size_t)QKVLD * DINc;            // 3,145,728
  const size_t nWout = (size_t)DOUTc * DOUTc;           // 1,048,576
  const size_t nQKV  = (size_t)Bc * Tc * QKVLD;         // 25,165,824
  const size_t nVT   = (size_t)Bc * NHc * HDc * Tc;     // 8,388,608
  const size_t nCTX  = (size_t)Bc * Tc * DOUTc;         // 8,388,608

  unsigned short* xb    = (unsigned short*)d_ws;
  unsigned short* wqkvb = xb + nX;
  unsigned short* woutb = wqkvb + nWqkv;
  unsigned short* qkvb  = woutb + nWout;
  unsigned short* vtb   = qkvb + nQKV;
  unsigned short* ctxb  = vtb + nVT;

  const int TPB = 256;

  // 1) convert inputs to bf16
  cvt_f32_bf16<<<2048, TPB, 0, stream>>>(x, xb, nX);
  cvt_f32_bf16<<<1024, TPB, 0, stream>>>(w_qkv, wqkvb, nWqkv);
  cvt_f32_bf16<<<512,  TPB, 0, stream>>>(w_out, woutb, nWout);

  // 2) qkv = x @ w_qkv^T   M=8192 N=3072 K=1024
  {
    const int M = Bc * Tc, N = QKVLD, K = DINc;
    const int waves = (M / 64) * (N / 64);
    gemm64x64<false><<<waves * 32 / TPB, TPB, 0, stream>>>(xb, wqkvb, qkvb, M, N, K);
  }

  // 3) V -> V^T
  transpose_v<<<2048, TPB, 0, stream>>>(qkvb, vtb);

  // 4) attention
  {
    const int waves = Bc * NHc * (Tc / 16);              // 8192
    attn_kernel<<<waves * 32 / TPB, TPB, 0, stream>>>(qkvb, vtb, ctxb);
  }

  // 5) out = ctx @ w_out^T  M=8192 N=1024 K=1024 (fp32 output)
  {
    const int M = Bc * Tc, N = DOUTc, K = DOUTc;
    const int waves = (M / 64) * (N / 64);
    gemm64x64<true><<<waves * 32 / TPB, TPB, 0, stream>>>(ctxb, woutb, out, M, N, K);
  }
}
